// DecoderRNN_90881507983491
// MI455X (gfx1250) — matrix operand order repaired and checked by reading, not compiled, not run on hardware
//
#include <hip/hip_runtime.h>
#include <hip/hip_bf16.h>

// ---------------- CDNA5 (gfx1250) types for WMMA ----------------
typedef __bf16        v16bf __attribute__((ext_vector_type(16)));
typedef float         v8f   __attribute__((ext_vector_type(8)));
typedef unsigned int  u32x4 __attribute__((ext_vector_type(4)));
typedef unsigned int  u32x2 __attribute__((ext_vector_type(2)));
typedef float         f32x4 __attribute__((ext_vector_type(4)));

#define HDIM 2048
#define BDIM 256
#define TDIM 512
#define ODIM 14
#define G3H  6144   // 3*HDIM

// fp32 -> bf16 round-to-nearest-even
__device__ __forceinline__ unsigned short f2bf(float f) {
    unsigned int u = __float_as_uint(f);
    unsigned int r = u + 0x7FFFu + ((u >> 16) & 1u);
    return (unsigned short)(r >> 16);
}

// ---------------- one-time precompute: W_hh fp32 -> bf16 ----------------
__global__ __launch_bounds__(256) void k_cvt_whh(const float* __restrict__ w,
                                                 unsigned short* __restrict__ out) {
    size_t i = ((size_t)blockIdx.x * 256 + threadIdx.x) * 4;   // 3H*H divisible by 4
    if (i + 3 < (size_t)G3H * HDIM) {
        f32x4 v = *(const f32x4*)(w + i);
        u32x2 p;
        p.x = (unsigned int)f2bf(v.x) | ((unsigned int)f2bf(v.y) << 16);
        p.y = (unsigned int)f2bf(v.z) | ((unsigned int)f2bf(v.w) << 16);
        *(u32x2*)(out + i) = p;
    }
}

// ---------------- one-time init: h0, h0_bf16, x0 ----------------
__global__ __launch_bounds__(256) void k_init(const float* __restrict__ enc_hid,  // (1,B,H)
                                              const float* __restrict__ tgt,      // (B,T,O)
                                              float* __restrict__ h,
                                              unsigned short* __restrict__ hbf,
                                              float* __restrict__ x) {
    size_t i = (size_t)blockIdx.x * 256 + threadIdx.x;
    if (i < (size_t)BDIM * HDIM) {
        float v = enc_hid[i];
        h[i]   = v;
        hbf[i] = f2bf(v);
    }
    if (i < (size_t)BDIM * ODIM) {
        size_t b = i / ODIM, o = i % ODIM;
        x[i] = tgt[b * (size_t)TDIM * ODIM + o];   // target_tensor[:, 0]
    }
}

// ---------------- per-step GEMM: gh = h @ W_hh^T + b_hh  (bf16 WMMA, f32 acc) ---------
// grid = (G3H/256, BDIM/32), block = 256 (8 waves). Each wave: 2x2 grid of 16x16 tiles.
// W_hh (25 MB bf16) is L2-resident after step 0; the clause-batched b128 demand loads
// (8 per k-step, issued ahead of the staged s_wait_loadcnt) keep the XDL pipe fed.
__global__ __launch_bounds__(256) void k_gemm_gh(const unsigned short* __restrict__ hbf,  // B x H
                                                 const unsigned short* __restrict__ wbf,  // 3H x H
                                                 const float* __restrict__ b_hh,          // 3H
                                                 float* __restrict__ gh) {                // B x 3H
    const int lane = threadIdx.x & 31;
    const int wave = threadIdx.x >> 5;
    const int half = lane >> 4;      // 0: lanes 0-15, 1: lanes 16-31
    const int l16  = lane & 15;

    const int m0 = blockIdx.y * 32;                 // two 16-row M tiles
    const int n0 = blockIdx.x * 256 + wave * 32;    // two 16-col N tiles

    // A-matrix (16x32 bf16) per-lane base: row = m0+l16, chunks at k+half*8 and k+16+half*8
    const unsigned short* a0p = hbf + (size_t)(m0 + l16) * HDIM + half * 8;
    const unsigned short* a1p = a0p + (size_t)16 * HDIM;
    // B-matrix (32x16 bf16): column n = row n of W_hh; chunks at k+half*8 and k+16+half*8
    const unsigned short* b0p = wbf + (size_t)(n0 + l16) * HDIM + half * 8;
    const unsigned short* b1p = b0p + (size_t)16 * HDIM;

    v8f c00 = {}, c01 = {}, c10 = {}, c11 = {};

    union Frag { u32x4 q[2]; v16bf v; };

    #pragma unroll 2
    for (int k = 0; k < HDIM; k += 32) {
        Frag a0, a1, b0, b1;
        a0.q[0] = *(const u32x4*)(a0p + k);
        a0.q[1] = *(const u32x4*)(a0p + k + 16);
        a1.q[0] = *(const u32x4*)(a1p + k);
        a1.q[1] = *(const u32x4*)(a1p + k + 16);
        b0.q[0] = *(const u32x4*)(b0p + k);
        b0.q[1] = *(const u32x4*)(b0p + k + 16);
        b1.q[0] = *(const u32x4*)(b1p + k);
        b1.q[1] = *(const u32x4*)(b1p + k + 16);

        c00 = __builtin_amdgcn_wmma_f32_16x16x32_bf16(false, a0.v, false, b0.v, (short)0, c00, false, false);
        c01 = __builtin_amdgcn_wmma_f32_16x16x32_bf16(false, a0.v, false, b1.v, (short)0, c01, false, false);
        c10 = __builtin_amdgcn_wmma_f32_16x16x32_bf16(false, a1.v, false, b0.v, (short)0, c10, false, false);
        c11 = __builtin_amdgcn_wmma_f32_16x16x32_bf16(false, a1.v, false, b1.v, (short)0, c11, false, false);
    }

    // C/D layout: VGPR v -> M = v + 8*half, N = l16
    const float bias0 = b_hh[n0 + l16];
    const float bias1 = b_hh[n0 + 16 + l16];
    const int   rb    = m0 + half * 8;
    #pragma unroll
    for (int v = 0; v < 8; ++v) {
        const size_t r0 = (size_t)(rb + v) * G3H;
        const size_t r1 = (size_t)(rb + v + 16) * G3H;
        gh[r0 + n0 + l16]      = c00[v] + bias0;
        gh[r0 + n0 + 16 + l16] = c01[v] + bias1;
        gh[r1 + n0 + l16]      = c10[v] + bias0;
        gh[r1 + n0 + 16 + l16] = c11[v] + bias1;
    }
}

// ---------------- per-step GRU gating: h_new = (1-z)*n + z*h ----------------
__global__ __launch_bounds__(256) void k_gate(const float* __restrict__ gh,
                                              const float* __restrict__ x,     // B x 14
                                              const float* __restrict__ W_ih,  // 3H x 14
                                              const float* __restrict__ b_ih,  // 3H
                                              float* __restrict__ h,
                                              unsigned short* __restrict__ hbf) {
    const size_t i = (size_t)blockIdx.x * 256 + threadIdx.x;   // i in [0, B*H)
    const int b = (int)(i >> 11);
    const int j = (int)(i & (HDIM - 1));

    float xr[ODIM];
    const float* xb = x + (size_t)b * ODIM;
    #pragma unroll
    for (int k = 0; k < ODIM; ++k) xr[k] = fmaxf(xb[k], 0.0f);

    const float* w0 = W_ih + (size_t)j * ODIM;
    const float* w1 = W_ih + (size_t)(j + HDIM) * ODIM;
    const float* w2 = W_ih + (size_t)(j + 2 * HDIM) * ODIM;
    float gir = b_ih[j], giz = b_ih[j + HDIM], gin = b_ih[j + 2 * HDIM];
    #pragma unroll
    for (int k = 0; k < ODIM; ++k) {
        gir = fmaf(xr[k], w0[k], gir);
        giz = fmaf(xr[k], w1[k], giz);
        gin = fmaf(xr[k], w2[k], gin);
    }

    const size_t gb = (size_t)b * G3H;
    const float ghr = gh[gb + j];
    const float ghz = gh[gb + HDIM + j];
    const float ghn = gh[gb + 2 * HDIM + j];

    const float r  = 1.0f / (1.0f + __expf(-(gir + ghr)));
    const float z  = 1.0f / (1.0f + __expf(-(giz + ghz)));
    const float n  = tanhf(gin + r * ghn);
    const float hv = h[i];
    const float hn = (1.0f - z) * n + z * hv;

    h[i]   = hn;
    hbf[i] = f2bf(hn);
}

// ---------------- per-step logits / argmax / log_softmax (block per batch row) --------
__global__ __launch_bounds__(256) void k_logits(const float* __restrict__ h,
                                                const float* __restrict__ W_out,  // 14 x H
                                                const float* __restrict__ b_out,  // 14
                                                float* __restrict__ log_probs,    // (B,T,O) in d_out
                                                int t, int* __restrict__ topi,
                                                int* __restrict__ hit) {
    const int b   = blockIdx.x;
    const int tid = threadIdx.x;
    __shared__ float red[ODIM][256];

    float acc[ODIM];
    #pragma unroll
    for (int o = 0; o < ODIM; ++o) acc[o] = 0.0f;

    const float* hb = h + (size_t)b * HDIM;
    for (int j = tid; j < HDIM; j += 256) {
        const float hv = hb[j];
        #pragma unroll
        for (int o = 0; o < ODIM; ++o) acc[o] = fmaf(hv, W_out[(size_t)o * HDIM + j], acc[o]);
    }
    #pragma unroll
    for (int o = 0; o < ODIM; ++o) red[o][tid] = acc[o];
    __syncthreads();
    for (int s = 128; s > 0; s >>= 1) {
        if (tid < s) {
            #pragma unroll
            for (int o = 0; o < ODIM; ++o) red[o][tid] += red[o][tid + s];
        }
        __syncthreads();
    }

    if (tid == 0) {
        float l[ODIM];
        float m = -3.4e38f;
        int am = 0;
        #pragma unroll
        for (int o = 0; o < ODIM; ++o) {
            l[o] = red[o][0] + b_out[o];
            if (l[o] > m) { m = l[o]; am = o; }
        }
        float s = 0.0f;
        #pragma unroll
        for (int o = 0; o < ODIM; ++o) s += __expf(l[o] - m);
        const float lse = __logf(s);
        float* lp = log_probs + (size_t)b * TDIM * ODIM + (size_t)t * ODIM;
        #pragma unroll
        for (int o = 0; o < ODIM; ++o) lp[o] = l[o] - m - lse;
        topi[b] = am;
        hit[b]  = (am == ODIM - 1) ? 1 : 0;
    }
}

// ---------------- per-step feedback: alive mask (inclusive cumsum) + one-hot ----------
__global__ __launch_bounds__(256) void k_feedback(const int* __restrict__ topi,
                                                  const int* __restrict__ hit,
                                                  float* __restrict__ x) {
    __shared__ float alive[BDIM];
    if (threadIdx.x == 0) {
        int c = 0;
        for (int b = 0; b < BDIM; ++b) {
            c += hit[b];                       // inclusive cumsum per reference
            alive[b] = (c == 0) ? 1.0f : 0.0f;
        }
    }
    __syncthreads();
    const int b  = threadIdx.x;
    const int tp = topi[b];
    const float a = alive[b];
    float* xb = x + (size_t)b * ODIM;
    #pragma unroll
    for (int o = 0; o < ODIM; ++o) xb[o] = (o == tp) ? a : 0.0f;
}

// ---------------- host-side orchestration ----------------
extern "C" void kernel_launch(void* const* d_in, const int* in_sizes, int n_in,
                              void* d_out, int out_size, void* d_ws, size_t ws_size,
                              hipStream_t stream) {
    const float* enc_hid = (const float*)d_in[1];   // (1,B,H)
    const float* tgt     = (const float*)d_in[2];   // (B,T,O)
    const float* W_ih    = (const float*)d_in[3];   // (3H,14)
    const float* W_hh    = (const float*)d_in[4];   // (3H,H)
    const float* b_ih    = (const float*)d_in[5];
    const float* b_hh    = (const float*)d_in[6];
    const float* W_out   = (const float*)d_in[7];   // (14,H)
    const float* b_out   = (const float*)d_in[8];
    float* out = (float*)d_out;

    // workspace carve-up (256B aligned)
    char*  ws  = (char*)d_ws;
    size_t off = 0;
    auto carve = [&](size_t bytes) -> void* {
        void* p = ws + off;
        off += (bytes + 255) & ~(size_t)255;
        return p;
    };
    unsigned short* whh_bf = (unsigned short*)carve((size_t)G3H * HDIM * 2);   // 25.2 MB
    float*          h      = (float*)         carve((size_t)BDIM * HDIM * 4);  //  2.1 MB
    unsigned short* hbf    = (unsigned short*)carve((size_t)BDIM * HDIM * 2);  //  1.0 MB
    float*          gh     = (float*)         carve((size_t)BDIM * G3H * 4);   //  6.3 MB
    float*          x      = (float*)         carve((size_t)BDIM * ODIM * 4);
    int*            topi   = (int*)           carve((size_t)BDIM * 4);
    int*            hit    = (int*)           carve((size_t)BDIM * 4);
    (void)ws_size; (void)in_sizes; (void)n_in; (void)out_size;

    // precompute: bf16 weights + initial state
    {
        const size_t n4 = (size_t)G3H * HDIM / 4;
        k_cvt_whh<<<dim3((unsigned)((n4 + 255) / 256)), dim3(256), 0, stream>>>(W_hh, whh_bf);
        const size_t ni = (size_t)BDIM * HDIM;
        k_init<<<dim3((unsigned)((ni + 255) / 256)), dim3(256), 0, stream>>>(enc_hid, tgt, h, hbf, x);
    }

    // sequential scan over T steps
    for (int t = 0; t < TDIM; ++t) {
        k_gemm_gh<<<dim3(G3H / 256, BDIM / 32), dim3(256), 0, stream>>>(hbf, whh_bf, b_hh, gh);
        k_gate<<<dim3((BDIM * HDIM) / 256), dim3(256), 0, stream>>>(gh, x, W_ih, b_ih, h, hbf);
        k_logits<<<dim3(BDIM), dim3(256), 0, stream>>>(h, W_out, b_out, out, t, topi, hit);
        k_feedback<<<dim3(1), dim3(256), 0, stream>>>(topi, hit, x);
    }

    // h_final -> tail of d_out
    hipMemcpyAsync(out + (size_t)BDIM * TDIM * ODIM, h,
                   (size_t)BDIM * HDIM * sizeof(float),
                   hipMemcpyDeviceToDevice, stream);
}
